// CNN_BiMAMBA_each_AV_42288247997050
// MI455X (gfx1250) — compile-verified, matrix operand order checked by reading, not run on hardware
//
#include <hip/hip_runtime.h>
#include <hip/hip_bf16.h>

typedef __attribute__((ext_vector_type(16))) _Float16 v16h;
typedef __attribute__((ext_vector_type(8)))  float    v8f;

// ---------------------------------------------------------------------------
// Implicit-GEMM 3x3 SAME conv with WMMA f16 (f32 accumulate).
// One wave computes a 32(pixels) x 32(cout) macro-tile: 2x2 grid of 16x16
// WMMA tiles, so each staged A tile feeds 2 WMMAs and each B tile feeds 2.
// POOL==1: fused ReLU + 2x2 maxpool (atomicMax on uint image of >=0 floats,
//          pooled buffer pre-zeroed).
// POOL==0: fused ReLU + global mean pool (atomicAdd of v/HW into feat[n][c]).
// ---------------------------------------------------------------------------
template<int CIN, int COUT, int H, int W, int POOL>
__global__ __launch_bounds__(32)
void conv3x3_wmma(const float* __restrict__ in, const float* __restrict__ wgt,
                  const float* __restrict__ bias, float* __restrict__ out)
{
  constexpr int K  = CIN * 9;
  constexpr int KS = (K + 31) / 32;
  constexpr int HW = H * W;

  __shared__ _Float16 As[2][16][32];
  __shared__ _Float16 Bs[2][16][32];

  const int l    = threadIdx.x;
  const int img  = blockIdx.z;
  const int m0   = blockIdx.x * 32;      // first pixel of macro-tile
  const int n0   = blockIdx.y * 32;      // first cout of macro-tile
  const int r    = l & 15;
  const int half = l >> 4;

  v8f acc00 = {}, acc01 = {}, acc10 = {}, acc11 = {};

  for (int ks = 0; ks < KS; ++ks) {
    const int kbase = ks * 32;
    __syncthreads();
    // Stage two A tiles (im2col rows for pixels m0..m0+31).
    #pragma unroll
    for (int mt = 0; mt < 2; ++mt) {
      const int p  = m0 + mt * 16 + r;
      const int py = p / W;
      const int px = p - py * W;
      #pragma unroll 4
      for (int j = 0; j < 16; ++j) {
        const int k = kbase + half * 16 + j;
        float av = 0.0f;
        if (k < K) {
          const int cin = k / 9;
          const int t   = k - cin * 9;
          const int ky  = t / 3;
          const int kx  = t - ky * 3;
          const int iy  = py + ky - 1;
          const int ix  = px + kx - 1;
          if (p < HW && iy >= 0 && iy < H && ix >= 0 && ix < W)
            av = in[((img * CIN + cin) * H + iy) * W + ix];
        }
        As[mt][r][half * 16 + j] = (_Float16)av;
      }
    }
    // Stage two B tiles (weights for couts n0..n0+31).
    #pragma unroll
    for (int nt = 0; nt < 2; ++nt) {
      const int co = n0 + nt * 16 + r;
      #pragma unroll 4
      for (int j = 0; j < 16; ++j) {
        const int k = kbase + half * 16 + j;
        Bs[nt][r][half * 16 + j] = (_Float16)((k < K) ? wgt[co * K + k] : 0.0f);
      }
    }
    __syncthreads();
    // Per-lane fragment gather per CDNA5 16-bit A/B layout:
    // lanes 0-15: K in {0..7} U {16..23}; lanes 16-31: K in {8..15} U {24..31}.
    const int kb = half * 8;
    v16h a0, a1, b0, b1;
    #pragma unroll
    for (int e = 0; e < 8; ++e) {
      a0[e] = As[0][r][kb + e];  a0[e + 8] = As[0][r][kb + 16 + e];
      a1[e] = As[1][r][kb + e];  a1[e + 8] = As[1][r][kb + 16 + e];
      b0[e] = Bs[0][r][kb + e];  b0[e + 8] = Bs[0][r][kb + 16 + e];
      b1[e] = Bs[1][r][kb + e];  b1[e + 8] = Bs[1][r][kb + 16 + e];
    }
    acc00 = __builtin_amdgcn_wmma_f32_16x16x32_f16(false, a0, false, b0, (short)0, acc00, false, false);
    acc01 = __builtin_amdgcn_wmma_f32_16x16x32_f16(false, a0, false, b1, (short)0, acc01, false, false);
    acc10 = __builtin_amdgcn_wmma_f32_16x16x32_f16(false, a1, false, b0, (short)0, acc10, false, false);
    acc11 = __builtin_amdgcn_wmma_f32_16x16x32_f16(false, a1, false, b1, (short)0, acc11, false, false);
  }

  // Epilogue. D layout: N = lane&15, M = 8*(lane>>4) + vgpr.
  auto epi = [&](const v8f& a, int mt, int nt) {
    const int cout = n0 + nt * 16 + r;
    const float bia = bias[cout];
    #pragma unroll
    for (int rr = 0; rr < 8; ++rr) {
      const int pp = m0 + mt * 16 + half * 8 + rr;
      if (pp >= HW) continue;
      const int yy = pp / W;
      const int xx = pp - yy * W;
      float v = a[rr] + bia;
      v = v > 0.0f ? v : 0.0f;             // ReLU (=> v >= 0)
      if (POOL == 1) {
        // 2x2 maxpool: uint compare == float compare for non-negative floats.
        unsigned int* dst = (unsigned int*)
          &out[((size_t)(img * COUT + cout) * (H / 2) + (yy >> 1)) * (W / 2) + (xx >> 1)];
        atomicMax(dst, __float_as_uint(v));
      } else {
        atomicAdd(&out[img * COUT + cout], v * (1.0f / (float)HW));
      }
    }
  };
  epi(acc00, 0, 0); epi(acc01, 0, 1); epi(acc10, 1, 0); epi(acc11, 1, 1);
}

// ---------------------------------------------------------------------------
// Small fp32 helper kernels (Mamba stacks + heads). These shapes (K=13, N=45,
// sequential scan over L) don't profit from 16x16 WMMA tiles.
// ---------------------------------------------------------------------------
__device__ __forceinline__ float silu_f(float x) { return x / (1.0f + expf(-x)); }
__device__ __forceinline__ float softplus_f(float x) {
  return x > 20.0f ? x : log1pf(expf(x));
}

__global__ void rmsnorm_k(const float* __restrict__ x, const float* __restrict__ w,
                          float* __restrict__ out, int d)
{
  const int row = blockIdx.x;            // 256 rows (B*L)
  __shared__ float red[256];
  float s = 0.0f;
  for (int i = threadIdx.x; i < d; i += blockDim.x) {
    float v = x[row * d + i];
    s += v * v;
  }
  red[threadIdx.x] = s;
  __syncthreads();
  for (int off = 128; off > 0; off >>= 1) {
    if ((int)threadIdx.x < off) red[threadIdx.x] += red[threadIdx.x + off];
    __syncthreads();
  }
  const float inv = rsqrtf(red[0] / (float)d + 1e-5f);
  for (int i = threadIdx.x; i < d; i += blockDim.x)
    out[row * d + i] = x[row * d + i] * inv * w[i];
}

// out[m*ldo+n] = f( dot(in[m*lda+0..K), W[n*K+0..K)) + bias[n]? ) (+ resid)
// mode 0: plain; 1: softplus(.+bias); 2: + resid[m*ldo+n]
__global__ void gemm_k(const float* __restrict__ in, int lda,
                       const float* __restrict__ W, const float* __restrict__ bias,
                       const float* __restrict__ resid, float* __restrict__ out,
                       int ldo, int M, int N, int Kd, int mode)
{
  const int t = blockIdx.x * blockDim.x + threadIdx.x;
  if (t >= M * N) return;
  const int m = t / N;
  const int n = t - m * N;
  const float* a = in + (size_t)m * lda;
  const float* b = W + (size_t)n * Kd;
  float s = 0.0f;
  for (int k = 0; k < Kd; ++k) s += a[k] * b[k];
  if (mode == 1) s = softplus_f(s + bias[n]);
  else if (mode == 2) s += resid[(size_t)m * ldo + n];
  out[(size_t)m * ldo + n] = s;
}

// causal depthwise conv (k=4) over L on the x-half of xz, then SiLU.
__global__ void convsilu_k(const float* __restrict__ xz, const float* __restrict__ cw,
                           const float* __restrict__ cb, float* __restrict__ xi,
                           int L, int ed, int total)
{
  const int idx = blockIdx.x * blockDim.x + threadIdx.x;
  if (idx >= total) return;
  const int e   = idx % ed;
  const int bl  = idx / ed;              // b*L + l
  const int l   = bl % L;
  const int b   = bl / L;
  float s = cb[e];
  #pragma unroll
  for (int k = 0; k < 4; ++k) {
    const int ll = l - 3 + k;
    if (ll >= 0)
      s += xz[((size_t)(b * L + ll)) * (2 * ed) + e] * cw[e * 4 + k];
  }
  xi[idx] = silu_f(s);
}

// Selective-state scan: one thread per (b, e); 16-wide state, L sequential.
__global__ void ssm_k(const float* __restrict__ delta, const float* __restrict__ dbc,
                      int dtr, const float* __restrict__ xi,
                      const float* __restrict__ A_log, const float* __restrict__ Dp,
                      float* __restrict__ y, int L, int ed, int B)
{
  const int t = blockIdx.x * blockDim.x + threadIdx.x;
  if (t >= B * ed) return;
  const int e = t % ed;
  const int b = t / ed;
  const int nx = dtr + 32;
  float A[16], h[16];
  #pragma unroll
  for (int n = 0; n < 16; ++n) { A[n] = -expf(A_log[e * 16 + n]); h[n] = 0.0f; }
  const float De = Dp[e];
  for (int l = 0; l < L; ++l) {
    const size_t row = (size_t)(b * L + l);
    const float dl = delta[row * ed + e];
    const float xv = xi[row * ed + e];
    const float* bc = dbc + row * nx;
    float acc = 0.0f;
    #pragma unroll
    for (int n = 0; n < 16; ++n) {
      h[n] = expf(dl * A[n]) * h[n] + dl * bc[dtr + n] * xv;
      acc += h[n] * bc[dtr + 16 + n];
    }
    y[row * ed + e] = acc + De * xv;
  }
}

__global__ void gate_k(float* __restrict__ y, const float* __restrict__ xz,
                       int ed, int total)
{
  const int idx = blockIdx.x * blockDim.x + threadIdx.x;
  if (idx >= total) return;
  const int e = idx % ed;
  const int m = idx / ed;
  const float z = xz[(size_t)m * (2 * ed) + ed + e];
  y[idx] *= silu_f(z);
}

// CNN fc: feat(256,128) -> xv(B,L,196); also writes time-flipped copy for vb.
__global__ void fc_cnn_k(const float* __restrict__ feat, const float* __restrict__ w,
                         const float* __restrict__ b, float* __restrict__ xv,
                         float* __restrict__ xvb)
{
  const int t = blockIdx.x * blockDim.x + threadIdx.x;
  if (t >= 256 * 196) return;
  const int j = t % 196;
  const int m = t / 196;                  // b*64 + l
  const int bb = m / 64, l = m % 64;
  const float* a = feat + (size_t)m * 128;
  const float* ww = w + (size_t)j * 128;
  float s = b[j];
  for (int k = 0; k < 128; ++k) s += a[k] * ww[k];
  xv [(size_t)(bb * 64 + l)        * 196 + j] = s;
  xvb[(size_t)(bb * 64 + (63 - l)) * 196 + j] = s;
}

__global__ void prep_audio_k(const float* __restrict__ audio, float* __restrict__ xa,
                             float* __restrict__ xab)
{
  const int t = blockIdx.x * blockDim.x + threadIdx.x;
  if (t >= 256 * 60) return;
  const int c = t % 60;
  const int m = t / 60;                   // b*64 + l
  const int bb = m / 64, l = m % 64;
  const float v = audio[(size_t)bb * 3840 + l * 60 + c];
  xa [(size_t)(bb * 64 + l)        * 60 + c] = v;
  xab[(size_t)(bb * 64 + (63 - l)) * 60 + c] = v;
}

// xlast[b] = concat(vf[b,L-1], vb[b,L-1]=mamba(flip)[b,0], af[b,L-1], ab[b,0])
__global__ void concat_k(const float* __restrict__ xv, const float* __restrict__ xvb,
                         const float* __restrict__ xa, const float* __restrict__ xab,
                         float* __restrict__ xlast)
{
  const int b = blockIdx.x;
  const int j = threadIdx.x;              // 0..511
  float v;
  if (j < 196)      v = xv [(size_t)(b * 64 + 63) * 196 + j];
  else if (j < 392) v = xvb[(size_t)(b * 64 + 0)  * 196 + (j - 196)];
  else if (j < 452) v = xa [(size_t)(b * 64 + 63) * 60  + (j - 392)];
  else              v = xab[(size_t)(b * 64 + 0)  * 60  + (j - 452)];
  xlast[(size_t)b * 512 + j] = v;
}

__global__ void fc1_k(const float* __restrict__ xlast, const float* __restrict__ w,
                      const float* __restrict__ b, float* __restrict__ h1)
{
  const int bb = blockIdx.x;
  const int j  = threadIdx.x;             // 0..255
  const float* a  = xlast + (size_t)bb * 512;
  const float* ww = w + (size_t)j * 512;
  float s = b[j];
  for (int k = 0; k < 512; ++k) s += a[k] * ww[k];
  h1[(size_t)bb * 256 + j] = s;
}

__global__ void fc2_k(const float* __restrict__ h1, const float* __restrict__ w,
                      const float* __restrict__ b, float* __restrict__ out)
{
  const int t = threadIdx.x;              // 0..7
  if (t >= 8) return;
  const int bb = t >> 1, kk = t & 1;
  const float* a  = h1 + (size_t)bb * 256;
  const float* ww = w + (size_t)kk * 256;
  float s = b[kk];
  for (int j = 0; j < 256; ++j) s += a[j] * ww[j];
  out[bb * 2 + kk] = s;
}

// ---------------------------------------------------------------------------
// Host orchestration
// ---------------------------------------------------------------------------
struct MambaP {
  const float *norm_w, *in_proj_w, *conv_w, *conv_b, *x_proj_w,
              *dt_proj_w, *dt_proj_b, *A_log, *D, *out_proj_w;
};

static MambaP layer_at(void* const* din, int base, bool sorted) {
  const float* p[10];
  for (int i = 0; i < 10; ++i) p[i] = (const float*)din[base + i];
  MambaP m;
  if (!sorted) { // insertion order: norm,in_proj,conv_w,conv_b,x_proj,dt_w,dt_b,A_log,D,out_proj
    m.norm_w = p[0]; m.in_proj_w = p[1]; m.conv_w = p[2]; m.conv_b = p[3];
    m.x_proj_w = p[4]; m.dt_proj_w = p[5]; m.dt_proj_b = p[6];
    m.A_log = p[7]; m.D = p[8]; m.out_proj_w = p[9];
  } else {       // sorted keys: A_log,D,conv_b,conv_w,dt_b,dt_w,in_proj,norm,out_proj,x_proj
    m.A_log = p[0]; m.D = p[1]; m.conv_b = p[2]; m.conv_w = p[3];
    m.dt_proj_b = p[4]; m.dt_proj_w = p[5]; m.in_proj_w = p[6];
    m.norm_w = p[7]; m.out_proj_w = p[8]; m.x_proj_w = p[9];
  }
  return m;
}

static void run_stack(const MambaP* L2, float* x, int d, int ed, int dtr,
                      float* xn, float* xz, float* xi, float* dbc,
                      float* delta, float* yb, hipStream_t s)
{
  const int M = 256;                      // B*L
  const int nx = dtr + 32;
  for (int li = 0; li < 2; ++li) {
    const MambaP& P = L2[li];
    rmsnorm_k<<<M, 256, 0, s>>>(x, P.norm_w, xn, d);
    int n;
    n = M * 2 * ed;
    gemm_k<<<(n + 255) / 256, 256, 0, s>>>(xn, d, P.in_proj_w, nullptr, nullptr,
                                           xz, 2 * ed, M, 2 * ed, d, 0);
    n = M * ed;
    convsilu_k<<<(n + 255) / 256, 256, 0, s>>>(xz, P.conv_w, P.conv_b, xi, 64, ed, n);
    n = M * nx;
    gemm_k<<<(n + 255) / 256, 256, 0, s>>>(xi, ed, P.x_proj_w, nullptr, nullptr,
                                           dbc, nx, M, nx, ed, 0);
    n = M * ed;
    gemm_k<<<(n + 255) / 256, 256, 0, s>>>(dbc, nx, P.dt_proj_w, P.dt_proj_b, nullptr,
                                           delta, ed, M, ed, dtr, 1);
    n = 4 * ed;
    ssm_k<<<(n + 63) / 64, 64, 0, s>>>(delta, dbc, dtr, xi, P.A_log, P.D, yb, 64, ed, 4);
    n = M * ed;
    gate_k<<<(n + 255) / 256, 256, 0, s>>>(yb, xz, ed, n);
    n = M * d;
    gemm_k<<<(n + 255) / 256, 256, 0, s>>>(yb, ed, P.out_proj_w, nullptr, x,
                                           x, d, M, d, ed, 2);
  }
}

extern "C" void kernel_launch(void* const* d_in, const int* in_sizes, int n_in,
                              void* d_out, int out_size, void* d_ws, size_t ws_size,
                              hipStream_t stream)
{
  (void)n_in; (void)out_size; (void)ws_size;
  // Detect flattening convention: insertion order puts video (9,633,792 elems) first.
  const bool sorted = (in_sizes[0] != 9633792);

  const float *video, *audio, *w1, *b1, *w2, *b2, *w3, *b3, *cfw, *cfb;
  const float *tfw, *tfb, *f2w, *f2b;
  MambaP vf[2], vb[2], af[2], ab[2];

  if (!sorted) {
    video = (const float*)d_in[0];  audio = (const float*)d_in[1];
    w1 = (const float*)d_in[2];  b1 = (const float*)d_in[3];
    w2 = (const float*)d_in[4];  b2 = (const float*)d_in[5];
    w3 = (const float*)d_in[6];  b3 = (const float*)d_in[7];
    cfw = (const float*)d_in[8]; cfb = (const float*)d_in[9];
    vf[0] = layer_at(d_in, 10, false); vf[1] = layer_at(d_in, 20, false);
    vb[0] = layer_at(d_in, 30, false); vb[1] = layer_at(d_in, 40, false);
    af[0] = layer_at(d_in, 50, false); af[1] = layer_at(d_in, 60, false);
    ab[0] = layer_at(d_in, 70, false); ab[1] = layer_at(d_in, 80, false);
    tfw = (const float*)d_in[90]; tfb = (const float*)d_in[91];
    f2w = (const float*)d_in[92]; f2b = (const float*)d_in[93];
  } else {
    audio = (const float*)d_in[0];
    ab[0] = layer_at(d_in, 1, true);  ab[1] = layer_at(d_in, 11, true);
    af[0] = layer_at(d_in, 21, true); af[1] = layer_at(d_in, 31, true);
    b1 = (const float*)d_in[41]; b2 = (const float*)d_in[42]; b3 = (const float*)d_in[43];
    cfb = (const float*)d_in[44]; cfw = (const float*)d_in[45];
    w1 = (const float*)d_in[46]; w2 = (const float*)d_in[47]; w3 = (const float*)d_in[48];
    f2b = (const float*)d_in[49]; f2w = (const float*)d_in[50];
    tfb = (const float*)d_in[51]; tfw = (const float*)d_in[52];
    vb[0] = layer_at(d_in, 53, true); vb[1] = layer_at(d_in, 63, true);
    vf[0] = layer_at(d_in, 73, true); vf[1] = layer_at(d_in, 83, true);
    video = (const float*)d_in[93];
  }

  // Workspace carve-up (256B aligned).
  char* ws = (char*)d_ws;
  size_t off = 0;
  auto carve = [&](size_t bytes) -> float* {
    float* p = (float*)(ws + off);
    off = (off + bytes + 255) & ~(size_t)255;
    return p;
  };
  const size_t pool1_b = 256UL * 32 * 56 * 56 * 4;   // 102.8 MB
  const size_t pool2_b = 256UL * 64 * 28 * 28 * 4;   //  51.4 MB
  const size_t feat_b  = 256UL * 128 * 4;
  float* pool1 = carve(pool1_b);
  float* pool2 = carve(pool2_b);
  float* feat  = carve(feat_b);
  float* xv    = carve(256UL * 196 * 4);
  float* xvb   = carve(256UL * 196 * 4);
  float* xa    = carve(256UL * 60 * 4);
  float* xab   = carve(256UL * 60 * 4);
  float* xn    = carve(256UL * 196 * 4);
  float* xz    = carve(256UL * 784 * 4);
  float* xi    = carve(256UL * 392 * 4);
  float* dbc   = carve(256UL * 45 * 4);
  float* delta = carve(256UL * 392 * 4);
  float* yb    = carve(256UL * 392 * 4);
  float* xlast = carve(4UL * 512 * 4);
  float* h1    = carve(4UL * 256 * 4);

  // Atomic accumulators must start at zero every call.
  hipMemsetAsync(pool1, 0, pool1_b, stream);
  hipMemsetAsync(pool2, 0, pool2_b, stream);
  hipMemsetAsync(feat,  0, feat_b,  stream);

  // --- CNN (WMMA implicit GEMM, 32x32 macro-tiles, fused ReLU + pooling) ---
  // conv1: HW=12544 -> 392 m-tiles; COUT=32 -> 1 n-tile of 32.
  conv3x3_wmma<3, 32, 112, 112, 1><<<dim3(392, 1, 256), 32, 0, stream>>>(video, w1, b1, pool1);
  // conv2: HW=3136 -> 98 m-tiles; COUT=64 -> 2 n-tiles.
  conv3x3_wmma<32, 64, 56, 56, 1><<<dim3(98, 2, 256), 32, 0, stream>>>(pool1, w2, b2, pool2);
  // conv3: HW=784 -> 25 m-tiles (last partially masked); COUT=128 -> 4 n-tiles.
  conv3x3_wmma<64, 128, 28, 28, 0><<<dim3(25, 4, 256), 32, 0, stream>>>(pool2, w3, b3, feat);

  fc_cnn_k<<<(256 * 196 + 255) / 256, 256, 0, stream>>>(feat, cfw, cfb, xv, xvb);
  prep_audio_k<<<(256 * 60 + 255) / 256, 256, 0, stream>>>(audio, xa, xab);

  // --- Bi-directional Mamba stacks (fp32 VALU; shapes too small for WMMA) ---
  run_stack(vf, xv,  196, 392, 13, xn, xz, xi, dbc, delta, yb, stream);
  run_stack(vb, xvb, 196, 392, 13, xn, xz, xi, dbc, delta, yb, stream);
  run_stack(af, xa,   60, 120,  4, xn, xz, xi, dbc, delta, yb, stream);
  run_stack(ab, xab,  60, 120,  4, xn, xz, xi, dbc, delta, yb, stream);

  // --- Heads ---
  concat_k<<<4, 512, 0, stream>>>(xv, xvb, xa, xab, xlast);
  fc1_k<<<4, 256, 0, stream>>>(xlast, tfw, tfb, h1);
  fc2_k<<<1, 8, 0, stream>>>(h1, f2w, f2b, (float*)d_out);
}
